// ProbSparseAttention_33887291965463
// MI455X (gfx1250) — matrix operand order repaired and checked by reading, not compiled, not run on hardware
//
#include <hip/hip_runtime.h>
#include <hip/hip_bf16.h>
#include <math.h>

#define B_    8
#define LQ_   4096
#define LK_   32768
#define D_    64
#define LKS_  8000
#define U_    45          // u = min(5*ceil(log1p(4096)), 4096) = 45
#define UT_   3           // ceil(45/16) row tiles
#define NSPLIT 64
#define KEYS_PER_SPLIT (LK_ / NSPLIT)   // 512

typedef __attribute__((ext_vector_type(16))) __bf16 v16bf;
typedef __attribute__((ext_vector_type(8)))  __bf16 v8bf;
typedef __attribute__((ext_vector_type(4)))  __bf16 v4bf;
typedef __attribute__((ext_vector_type(8)))  float  v8f;

// --- k-order maps used by this code (16-bit tiles, wave32; contiguous, not ISA table order) ---
// A 16x32: lane = row M (lane&15); element e <-> K = 16h + e
//          (h = lane>>4)
// B 32x16: lane = col N (lane&15); element e <-> K = 16h + e
__device__ __forceinline__ int kAmap(int j, int h) {
  return 16*h + 2*j;
}
__device__ __forceinline__ int kBmap(int j, int h) { return 16*h + 2*j; }

// --- bf16 direct tile loads (pre-packed operands) ---
__device__ __forceinline__ v16bf load_A_bf(const __bf16* rowp, int h) {
  v8bf lo = *reinterpret_cast<const v8bf*>(rowp + 16*h);       // K=16h..16h+7
  v8bf hi = *reinterpret_cast<const v8bf*>(rowp + 16*h + 8);   // K=16h+8..
  v16bf r;
#pragma unroll
  for (int e = 0; e < 8; ++e) { r[e] = lo[e]; r[8+e] = hi[e]; }
  return r;
}
__device__ __forceinline__ v16bf load_B_bf(const __bf16* rowp, int h) {
  return *reinterpret_cast<const v16bf*>(rowp + 16*h);         // K=16h..16h+15
}

// --- f32 tile loads with inline conversion (kernel 3, memory-bound) ---
__device__ __forceinline__ v16bf load_row_A(const float* rowp, int h) {
  v16bf r;
#pragma unroll
  for (int j = 0; j < 8; ++j) {
    int k0 = kAmap(j, h);
    float2 t = *reinterpret_cast<const float2*>(rowp + k0);
    r[2*j]   = (__bf16)t.x;
    r[2*j+1] = (__bf16)t.y;
  }
  return r;
}
__device__ __forceinline__ v16bf load_row_B(const float* rowp, int h) {
  v16bf r;
#pragma unroll
  for (int j = 0; j < 8; ++j) {
    int k0 = kBmap(j, h);
    float2 t = *reinterpret_cast<const float2*>(rowp + k0);
    r[2*j]   = (__bf16)t.x;
    r[2*j+1] = (__bf16)t.y;
  }
  return r;
}
__device__ __forceinline__ v16bf load_col_B(const float* base, int h, int stride) {
  v16bf r;
#pragma unroll
  for (int j = 0; j < 8; ++j) {
    int k0 = kBmap(j, h);
    r[2*j]   = (__bf16)base[(size_t)k0 * stride];
    r[2*j+1] = (__bf16)base[(size_t)(k0 + 1) * stride];
  }
  return r;
}

// ------------- Kernel 0a: pack q -> bf16 (row-major) ----------------------
__global__ __launch_bounds__(256)
void pack_q_kernel(const float* __restrict__ q, __bf16* __restrict__ qbf) {
  size_t i = ((size_t)blockIdx.x * 256 + threadIdx.x) * 4;
  float4 t = *reinterpret_cast<const float4*>(q + i);
  v4bf o;
  o[0] = (__bf16)t.x; o[1] = (__bf16)t.y; o[2] = (__bf16)t.z; o[3] = (__bf16)t.w;
  *reinterpret_cast<v4bf*>(qbf + i) = o;
}

// ------------- Kernel 0b: gather K[index_sample] -> bf16 ------------------
__global__ __launch_bounds__(256)
void pack_ksamp_kernel(const float* __restrict__ K, const int* __restrict__ idxs,
                       __bf16* __restrict__ ksbf) {
  size_t t = (size_t)blockIdx.x * 256 + threadIdx.x;   // B*LKS*16 threads
  int dg    = (int)(t & 15);                            // 4-dim group
  size_t row = t >> 4;                                  // b*LKS + u
  int b = (int)(row / LKS_);
  int u = (int)(row % LKS_);
  int ks = idxs[u];
  const float* src = K + ((size_t)b * LK_ + ks) * D_ + dg * 4;
  float4 v = *reinterpret_cast<const float4*>(src);
  v4bf o;
  o[0] = (__bf16)v.x; o[1] = (__bf16)v.y; o[2] = (__bf16)v.z; o[3] = (__bf16)v.w;
  *reinterpret_cast<v4bf*>(ksbf + row * D_ + dg * 4) = o;
}

// ---------------- Kernel 1: sampled QK + M = max - mean -------------------
__global__ __launch_bounds__(128)
void sampled_qk_kernel(const __bf16* __restrict__ qbf, const __bf16* __restrict__ ksbf,
                       float* __restrict__ Mv) {
  int lane = threadIdx.x & 31;
  int wid  = blockIdx.x * (blockDim.x >> 5) + (threadIdx.x >> 5);
  int b  = wid >> 8;            // 256 q-tiles per batch
  int qt = wid & 255;
  int h  = lane >> 4;
  int r  = lane & 15;

  const __bf16* qrow = qbf + ((size_t)b * LQ_ + qt * 16 + r) * D_;
  v16bf a0 = load_A_bf(qrow, h);
  v16bf a1 = load_A_bf(qrow + 32, h);

  float mx[8], sm[8];
#pragma unroll
  for (int v = 0; v < 8; ++v) { mx[v] = -3.0e38f; sm[v] = 0.f; }

  const __bf16* Ksb = ksbf + (size_t)b * LKS_ * D_;
  for (int u0 = 0; u0 < LKS_; u0 += 16) {
    const __bf16* krow = Ksb + (size_t)(u0 + r) * D_;   // lane's key column
    v16bf b0 = load_B_bf(krow, h);
    v16bf b1 = load_B_bf(krow + 32, h);
    v8f c = {};
    c = __builtin_amdgcn_wmma_f32_16x16x32_bf16(false, a0, false, b0, (short)0, c, false, false);
    c = __builtin_amdgcn_wmma_f32_16x16x32_bf16(false, a1, false, b1, (short)0, c, false, false);
#pragma unroll
    for (int v = 0; v < 8; ++v) { mx[v] = fmaxf(mx[v], c[v]); sm[v] += c[v]; }
  }
  // reduce across the 16 columns (stays inside 16-lane halves, wave32)
#pragma unroll
  for (int v = 0; v < 8; ++v) {
#pragma unroll
    for (int msk = 1; msk < 16; msk <<= 1) {
      mx[v] = fmaxf(mx[v], __shfl_xor(mx[v], msk, 32));
      sm[v] += __shfl_xor(sm[v], msk, 32);
    }
  }
  if (r == 0) {
#pragma unroll
    for (int v = 0; v < 8; ++v) {
      int row = v + 8 * h;
      Mv[(size_t)b * LQ_ + qt * 16 + row] = mx[v] - sm[v] * (1.0f / (float)LKS_);
    }
  }
}

// ---------------- Kernel 2: top-45 smallest M per batch -------------------
__global__ __launch_bounds__(256)
void topk_kernel(const float* __restrict__ Mv, int* __restrict__ topIdx) {
  __shared__ float vals[LQ_];
  __shared__ float redv[256];
  __shared__ int   redi[256];
  int b = blockIdx.x, t = threadIdx.x;
  for (int i = t; i < LQ_; i += 256) vals[i] = Mv[(size_t)b * LQ_ + i];
  __syncthreads();
  for (int it = 0; it < U_; ++it) {
    float bv = 3.0e38f; int bi = 0x7fffffff;
    for (int i = t; i < LQ_; i += 256) {
      float x = vals[i];
      if (x < bv) { bv = x; bi = i; }
    }
    redv[t] = bv; redi[t] = bi;
    __syncthreads();
    for (int s = 128; s > 0; s >>= 1) {
      if (t < s) {
        float ov = redv[t + s]; int oi = redi[t + s];
        if (ov < redv[t] || (ov == redv[t] && oi < redi[t])) { redv[t] = ov; redi[t] = oi; }
      }
      __syncthreads();
    }
    if (t == 0) { topIdx[b * U_ + it] = redi[0]; vals[redi[0]] = 3.0e38f; }
    __syncthreads();
  }
}

// ------- Kernel 3: scores (raw, to d_out) + split-K flash partials --------
__global__ __launch_bounds__(32)
void scores_pv_kernel(const float* __restrict__ q, const float* __restrict__ K,
                      const float* __restrict__ V, const int* __restrict__ topIdx,
                      float* __restrict__ scores_out,
                      float* __restrict__ pm, float* __restrict__ ps,
                      float* __restrict__ pacc) {
  __shared__ float m_lds[16];
  __shared__ float s_lds[16];
  __shared__ float p_lds[16][33];

  int lane = threadIdx.x;
  int h = lane >> 4, col = lane & 15, r = col;
  int blk = blockIdx.x;
  int sp = blk % NSPLIT;
  int ut = (blk / NSPLIT) % UT_;
  int b  = blk / (NSPLIT * UT_);

  int u = ut * 16 + r;
  int uq = topIdx[b * U_ + (u < U_ ? u : U_ - 1)];       // pad rows replicate row 44
  const float* qrow = q + ((size_t)b * LQ_ + uq) * D_;
  v16bf a0 = load_row_A(qrow, h);
  v16bf a1 = load_row_A(qrow + 32, h);

  if (lane < 16) { m_lds[lane] = -3.0e38f; s_lds[lane] = 0.f; }
  v8f acc[4];
#pragma unroll
  for (int g = 0; g < 4; ++g)
#pragma unroll
    for (int v = 0; v < 8; ++v) acc[g][v] = 0.f;
  __syncthreads();

  const float* Kb = K + (size_t)b * LK_ * D_;
  const float* Vb = V + (size_t)b * LK_ * D_;
  int kbase = sp * KEYS_PER_SPLIT;
  const float scale = 0.125f;    // 1/sqrt(64)

  for (int c0k = 0; c0k < KEYS_PER_SPLIT; c0k += 32) {
    int kc = kbase + c0k;
    v8f c0 = {}, c1 = {};
    {
      const float* krow = Kb + (size_t)(kc + col) * D_;
      v16bf b0 = load_row_B(krow, h);
      v16bf b1 = load_row_B(krow + 32, h);
      c0 = __builtin_amdgcn_wmma_f32_16x16x32_bf16(false, a0, false, b0, (short)0, c0, false, false);
      c0 = __builtin_amdgcn_wmma_f32_16x16x32_bf16(false, a1, false, b1, (short)0, c0, false, false);
    }
    {
      const float* krow = Kb + (size_t)(kc + 16 + col) * D_;
      v16bf b0 = load_row_B(krow, h);
      v16bf b1 = load_row_B(krow + 32, h);
      c1 = __builtin_amdgcn_wmma_f32_16x16x32_bf16(false, a0, false, b0, (short)0, c1, false, false);
      c1 = __builtin_amdgcn_wmma_f32_16x16x32_bf16(false, a1, false, b1, (short)0, c1, false, false);
    }
#pragma unroll
    for (int v = 0; v < 8; ++v) { c0[v] *= scale; c1[v] *= scale; }

    // raw (pre-softmax) scores straight to d_out
#pragma unroll
    for (int v = 0; v < 8; ++v) {
      int row = v + 8 * h;
      int uu = ut * 16 + row;
      if (uu < U_) {
        size_t base = (size_t)(b * U_ + uu) * LK_;
        scores_out[base + kc + col]      = c0[v];
        scores_out[base + kc + 16 + col] = c1[v];
      }
    }

    // online softmax update (per C-layout row = v + 8h)
    float fac[8];
#pragma unroll
    for (int v = 0; v < 8; ++v) {
      int row = v + 8 * h;
      float mold = m_lds[row];
      float lm = fmaxf(c0[v], c1[v]);
#pragma unroll
      for (int msk = 1; msk < 16; msk <<= 1) lm = fmaxf(lm, __shfl_xor(lm, msk, 32));
      float mnew = fmaxf(mold, lm);
      fac[v] = __expf(mold - mnew);
      float p0 = __expf(c0[v] - mnew);
      float p1 = __expf(c1[v] - mnew);
      float lsum = p0 + p1;
#pragma unroll
      for (int msk = 1; msk < 16; msk <<= 1) lsum += __shfl_xor(lsum, msk, 32);
      if (col == 0) { m_lds[row] = mnew; s_lds[row] = s_lds[row] * fac[v] + lsum; }
      p_lds[row][col]      = p0;
      p_lds[row][16 + col] = p1;
    }
    __syncthreads();

    // rescale running output accumulators
#pragma unroll
    for (int g = 0; g < 4; ++g)
#pragma unroll
      for (int v = 0; v < 8; ++v) acc[g][v] *= fac[v];

    // re-stage P (C layout -> A layout) through LDS as bf16
    v16bf ap;
#pragma unroll
    for (int j = 0; j < 8; ++j) {
      int k0 = kAmap(j, h);
      ap[2*j]   = (__bf16)p_lds[r][k0];
      ap[2*j+1] = (__bf16)p_lds[r][k0 + 1];
    }
    // P(16x32) x V(32x64) via 4 WMMAs (dim groups of 16)
#pragma unroll
    for (int g = 0; g < 4; ++g) {
      const float* vbase = Vb + (size_t)kc * D_ + (16 * g + col);
      v16bf bv = load_col_B(vbase, h, D_);
      acc[g] = __builtin_amdgcn_wmma_f32_16x16x32_bf16(false, ap, false, bv, (short)0, acc[g], false, false);
    }
    __syncthreads();
  }

  // emit split partials
  int pbase = ((b * UT_ + ut) * NSPLIT + sp) * 16;
  if (lane < 16) { pm[pbase + lane] = m_lds[lane]; ps[pbase + lane] = s_lds[lane]; }
#pragma unroll
  for (int g = 0; g < 4; ++g)
#pragma unroll
    for (int v = 0; v < 8; ++v) {
      int row = v + 8 * h;
      pacc[(size_t)(pbase + row) * D_ + 16 * g + col] = acc[g][v];
    }
}

// ---------------- Kernel 4: combine split-K partials ----------------------
__global__ __launch_bounds__(64)
void combine_kernel(const float* __restrict__ pm, const float* __restrict__ ps,
                    const float* __restrict__ pacc, float* __restrict__ out) {
  int d  = threadIdx.x;
  int ut = blockIdx.x % UT_;
  int b  = blockIdx.x / UT_;
  for (int row = 0; row < 16; ++row) {
    int u = ut * 16 + row;
    if (u >= U_) break;
    int base = ((b * UT_ + ut) * NSPLIT) * 16 + row;
    float gm = -3.0e38f;
    for (int sp = 0; sp < NSPLIT; ++sp) gm = fmaxf(gm, pm[base + sp * 16]);
    float S = 0.f, o = 0.f;
    for (int sp = 0; sp < NSPLIT; ++sp) {
      float w = __expf(pm[base + sp * 16] - gm);
      S += ps[base + sp * 16] * w;
      o += pacc[(size_t)(base + sp * 16) * D_ + d] * w;
    }
    out[(size_t)(b * U_ + u) * D_ + d] = o / S;
  }
}

extern "C" void kernel_launch(void* const* d_in, const int* in_sizes, int n_in,
                              void* d_out, int out_size, void* d_ws, size_t ws_size,
                              hipStream_t stream) {
  const float* q   = (const float*)d_in[0];
  const float* K   = (const float*)d_in[1];
  const float* V   = (const float*)d_in[2];
  const int* idx   = (const int*)d_in[3];
  float* out       = (float*)d_out;                 // attn_output: B*U*D floats
  float* scores    = out + (size_t)B_ * U_ * D_;    // attn_scores: B*U*LK floats

  // workspace layout: bf16 operand packs first (128B-aligned sizes), then f32
  char* wsb = (char*)d_ws;
  __bf16* qbf  = (__bf16*)wsb;                                   // B*LQ*D bf16 = 4MB
  __bf16* ksbf = (__bf16*)(wsb + (size_t)B_ * LQ_ * D_ * 2);     // B*LKS*D bf16 = 8MB
  float* fws   = (float*)(wsb + (size_t)B_ * LQ_ * D_ * 2 + (size_t)B_ * LKS_ * D_ * 2);
  float* Mv    = fws;                                  // B*LQ = 32768
  int*   topIx = (int*)(fws + (size_t)B_ * LQ_);       // B*U ints (pad to 384)
  float* pm    = fws + (size_t)B_ * LQ_ + 384;         // B*UT*NSPLIT*16 = 24576
  float* ps    = pm + (size_t)B_ * UT_ * NSPLIT * 16;
  float* pacc  = ps + (size_t)B_ * UT_ * NSPLIT * 16;  // *64 dims = 1.57M floats

  // 0) one-pass bf16 packing (q) and gather+pack (K[index_sample])
  pack_q_kernel<<<(B_ * LQ_ * D_ / 4) / 256, 256, 0, stream>>>(q, qbf);
  pack_ksamp_kernel<<<(B_ * LKS_ * (D_ / 4)) / 256, 256, 0, stream>>>(K, idx, ksbf);
  // 1) sampled QK -> M stats.  8 batches * 256 q-tiles, 4 waves/block.
  sampled_qk_kernel<<<(B_ * (LQ_ / 16)) / 4, 128, 0, stream>>>(qbf, ksbf, Mv);
  // 2) per-batch top-45 (smallest M, ties -> lower index)
  topk_kernel<<<B_, 256, 0, stream>>>(Mv, topIx);
  // 3) raw scores + split-K flash partials
  scores_pv_kernel<<<B_ * UT_ * NSPLIT, 32, 0, stream>>>(q, K, V, topIx, scores,
                                                         pm, ps, pacc);
  // 4) combine splits -> attn_output
  combine_kernel<<<B_ * UT_, 64, 0, stream>>>(pm, ps, pacc, out);
}
